// Model_39676907884787
// MI455X (gfx1250) — matrix operand order repaired and checked by reading, not compiled, not run on hardware
//
#include <hip/hip_runtime.h>
#include <stdint.h>
#include <math.h>

typedef __attribute__((ext_vector_type(16))) _Float16 v16h;
typedef __attribute__((ext_vector_type(2)))  __fp16   v2fp;
typedef __attribute__((ext_vector_type(8)))  float    v8f;

#define B_SZ   8
#define H_SZ   16
#define S_LEN  1024
#define D_DIM  128
#define KT     64          // keys per streamed tile
#define QROWS  128         // query rows per block (8 waves x 16)
#define NWAVES 8
#define VPITCH 72          // padded key-pitch for transposed V tile (bank spread)
#define NTILES (S_LEN / KT)
#define DROP_T 429496729u  // floor(0.1 * 2^32): keep if hash >= T  (p_keep = 0.9)
#define INV_KEEP 1.1111111111111112f

union F16Frag { v16h v; uint4 u[2]; };
union PK      { v2fp h; uint32_t u; };

__device__ __forceinline__ uint32_t hash32(uint32_t x) {
    x ^= x >> 16; x *= 0x7feb352du;
    x ^= x >> 15; x *= 0x846ca68bu;
    x ^= x >> 16; return x;
}

__global__ __launch_bounds__(256)
void fa_fwd_kernel(const float* __restrict__ Q, const float* __restrict__ K,
                   const float* __restrict__ V, const float* __restrict__ ISC,
                   float* __restrict__ O)
{
    // double-buffered f16 tiles: 2*16KB (K) + 2*18KB (V^T) + 16KB (P) = 84KB
    __shared__ __align__(16) _Float16 Kl[2][KT][D_DIM];
    __shared__ __align__(16) _Float16 Vt[2][D_DIM][VPITCH];
    __shared__ __align__(16) _Float16 Pl[NWAVES][16][KT];

    const int tid  = threadIdx.x;
    const int wave = tid >> 5;
    const int lane = tid & 31;
    const int hl   = lane & 15;        // lane index within 16-lane half
    const int hi   = lane >> 4;        // which half of the wave
    const int bh   = blockIdx.y;
    const int qbase = blockIdx.x * QROWS;

    const size_t base = (size_t)bh * S_LEN * D_DIM;
    const float* Qb = Q + base;
    const float* Kb = K + base;
    const float* Vb = V + base;
    float*       Ob = O + base;

    // ---- Q A-fragments in registers: 16x32 f16, 4 chunks covering D=128 ----
    const int  qrow = qbase + wave * 16 + hl;     // A layout: row = lane%16
    const int  aoff = hi ? 8 : 0;
    const float* qp = Qb + (size_t)qrow * D_DIM;
    v16h qfrag[4];
    #pragma unroll
    for (int c = 0; c < 4; ++c) {
        const float* p0 = qp + 32 * c + aoff;
        const float* p1 = qp + 32 * c + 16 + aoff;
        #pragma unroll
        for (int e = 0; e < 8; ++e) qfrag[c][e]     = (_Float16)p0[e];
        #pragma unroll
        for (int e = 0; e < 8; ++e) qfrag[c][8 + e] = (_Float16)p1[e];
    }

    v8f acc[8];
    #pragma unroll
    for (int j = 0; j < 8; ++j) {
        #pragma unroll
        for (int e = 0; e < 8; ++e) acc[j][e] = 0.0f;
    }
    float mrow[8], lrow[8];
    #pragma unroll
    for (int r = 0; r < 8; ++r) { mrow[r] = -INFINITY; lrow[r] = 0.0f; }

    const int myrow0 = qbase + wave * 16;

    // ---- register prefetch state for the next K/V tile ----
    float4 kr[8];          // K: 4 dims of one key per entry
    float2 va[8], vb[8];   // V: 2 dims of keys 2kp and 2kp+1

    auto stage_load = [&](int kt0) {
        #pragma unroll
        for (int i = 0; i < 8; ++i) {
            int flat = (i * 256 + tid) * 4;
            int key  = flat >> 7;            // /128
            int dim  = flat & (D_DIM - 1);
            kr[i] = *(const float4*)(Kb + (size_t)(kt0 + key) * D_DIM + dim);
            int idx = i * 256 + tid;         // (key-pair, dim-pair) grid 32x64
            int kp  = idx >> 6;
            int dp  = idx & 63;
            va[i] = *(const float2*)(Vb + (size_t)(kt0 + 2*kp    ) * D_DIM + 2*dp);
            vb[i] = *(const float2*)(Vb + (size_t)(kt0 + 2*kp + 1) * D_DIM + 2*dp);
        }
    };

    auto stage_store = [&](int buf) {
        #pragma unroll
        for (int i = 0; i < 8; ++i) {
            int flat = (i * 256 + tid) * 4;
            int key  = flat >> 7;
            int dim  = flat & (D_DIM - 1);
            PK p0, p1;
            p0.h = __builtin_amdgcn_cvt_pkrtz(kr[i].x, kr[i].y);
            p1.h = __builtin_amdgcn_cvt_pkrtz(kr[i].z, kr[i].w);
            uint2 w; w.x = p0.u; w.y = p1.u;
            *(uint2*)&Kl[buf][key][dim] = w;
            int idx = i * 256 + tid;
            int kp  = idx >> 6;
            int dp  = idx & 63;
            PK q0, q1;  // pack two adjacent keys of one dim -> one b32 store
            q0.h = __builtin_amdgcn_cvt_pkrtz(va[i].x, vb[i].x);
            q1.h = __builtin_amdgcn_cvt_pkrtz(va[i].y, vb[i].y);
            *(uint32_t*)&Vt[buf][2*dp    ][2*kp] = q0.u;
            *(uint32_t*)&Vt[buf][2*dp + 1][2*kp] = q1.u;
        }
    };

    stage_load(0);   // prologue: tile 0 -> registers

    for (int t = 0; t < NTILES; ++t) {
        const int kt0 = t * KT;
        const int buf = t & 1;

        stage_store(buf);           // registers -> LDS buffer for this tile
        __syncthreads();            // stores visible; prior compute on buf done
        if (t + 1 < NTILES)
            stage_load(kt0 + KT);   // issue next tile's loads; overlap compute

        // per-column reciprocal of inv_scale_factor
        float rs[4];
        #pragma unroll
        for (int j = 0; j < 4; ++j) rs[j] = 1.0f / ISC[kt0 + j * 16 + hl];

        // ---- scores S = Q * K^T for 4 subtiles of 16 keys ----
        v8f sc[4];
        #pragma unroll
        for (int j = 0; j < 4; ++j) {
            v8f a;
            #pragma unroll
            for (int e = 0; e < 8; ++e) a[e] = 0.0f;
            const int key  = j * 16 + hl;
            const int half = hi ? 8 : 0;
            #pragma unroll
            for (int c = 0; c < 4; ++c) {
                F16Frag kb;
                kb.u[0] = *(const uint4*)&Kl[buf][key][32 * c + half];
                kb.u[1] = *(const uint4*)&Kl[buf][key][32 * c + half + 16];
                a = __builtin_amdgcn_wmma_f32_16x16x32_f16(
                        false, qfrag[c], false, kb.v, (short)0, a, false, false);
            }
            sc[j] = a * rs[j];
        }

        // ---- online softmax: row max across 4 subtiles + 16 lanes of row ----
        #pragma unroll
        for (int r = 0; r < 8; ++r) {
            float m = sc[0][r];
            m = fmaxf(m, sc[1][r]); m = fmaxf(m, sc[2][r]); m = fmaxf(m, sc[3][r]);
            #pragma unroll
            for (int d = 1; d < 16; d <<= 1)
                m = fmaxf(m, __shfl_xor(m, d, 32));
            float mn    = fmaxf(mrow[r], m);
            float alpha = __expf(mrow[r] - mn);
            mrow[r] = mn;
            lrow[r] *= alpha;
            #pragma unroll
            for (int j = 0; j < 8; ++j) acc[j][r] *= alpha;
        }

        // ---- probabilities + dropout, staged per-wave to LDS as f16 ----
        #pragma unroll
        for (int j = 0; j < 4; ++j) {
            const int col = kt0 + j * 16 + hl;
            #pragma unroll
            for (int r = 0; r < 8; ++r) {
                const int rr  = hi ? r + 8 : r;
                const int row = myrow0 + rr;
                float p = __expf(sc[j][r] - mrow[r]);
                lrow[r] += p;                         // denominator: pre-dropout
                uint32_t h = hash32((uint32_t)(bh * (S_LEN * S_LEN) +
                                               row * S_LEN + col));
                float pd = (h >= DROP_T) ? p * INV_KEEP : 0.0f;
                Pl[wave][rr][j * 16 + hl] = (_Float16)pd;
            }
        }

        // ---- PV: acc(16x128) += P(16x64) * V(64x128), K-chunks of 32 keys ----
        #pragma unroll
        for (int c = 0; c < 2; ++c) {
            F16Frag pf;   // A-fragment from this wave's own P stage
            pf.u[0] = *(const uint4*)&Pl[wave][hl][32 * c + aoff];
            pf.u[1] = *(const uint4*)&Pl[wave][hl][32 * c + 16 + aoff];
            const int half = hi ? 8 : 0;
            #pragma unroll
            for (int j = 0; j < 8; ++j) {
                F16Frag vf;  // B-fragment: table-order keys from transposed V
                vf.u[0] = *(const uint4*)&Vt[buf][j * 16 + hl][32 * c + half];
                vf.u[1] = *(const uint4*)&Vt[buf][j * 16 + hl][32 * c + half + 16];
                acc[j] = __builtin_amdgcn_wmma_f32_16x16x32_f16(
                             false, pf.v, false, vf.v, (short)0, acc[j],
                             false, false);
            }
        }
    }

    // ---- finalize: reduce l across the 16 lanes of each row, normalize ----
    #pragma unroll
    for (int r = 0; r < 8; ++r) {
        float l = lrow[r];
        #pragma unroll
        for (int d = 1; d < 16; d <<= 1) l += __shfl_xor(l, d, 32);
        const float invl = 1.0f / l;
        const int row = myrow0 + (hi ? r + 8 : r);
        #pragma unroll
        for (int j = 0; j < 8; ++j)
            Ob[(size_t)row * D_DIM + j * 16 + hl] = acc[j][r] * invl;
    }
}

extern "C" void kernel_launch(void* const* d_in, const int* in_sizes, int n_in,
                              void* d_out, int out_size, void* d_ws, size_t ws_size,
                              hipStream_t stream) {
    (void)in_sizes; (void)n_in; (void)out_size; (void)d_ws; (void)ws_size;
    const float* q   = (const float*)d_in[0];
    const float* k   = (const float*)d_in[1];
    const float* v   = (const float*)d_in[2];
    const float* isc = (const float*)d_in[3];
    float* out = (float*)d_out;
    dim3 grid(S_LEN / QROWS, B_SZ * H_SZ);
    fa_fwd_kernel<<<grid, dim3(256), 0, stream>>>(q, k, v, isc, out);
}
